// FeedForward_72808285602173
// MI455X (gfx1250) — compile-verified, matrix operand order checked by reading, not run on hardware
//
#include <hip/hip_runtime.h>
#include <math.h>

// ---------------------------------------------------------------------------
// Equivariant FFN for MI455X (gfx1250, wave32, WMMA).
//   x:[N,480] -> LN -> per-l linear (480->960, bf16 WMMA, f32 acc) -> gate
//     -> per-l linear (960->480) -> x + y
// Internal LDS activation layouts are m-major per node so that:
//   * GEMM rows are ordered flat = m*16 + node  ->  rowtile == m, node == lane
//     (no integer division anywhere in the GEMM path)
//   * every A/B fragment is two contiguous 16-byte loads (ds/global b128)
// Tile dispatch runs on scalar (wave-uniform) control flow so EXEC is
// provably all-ones at every WMMA, as the ISA requires.
// ---------------------------------------------------------------------------

typedef __attribute__((ext_vector_type(16))) __bf16 v16bf;
typedef __attribute__((ext_vector_type(8)))  __bf16 v8bf;
typedef __attribute__((ext_vector_type(8)))  float  v8f;

#define XROW 480      // node feature width: 128 + 64*3 + 32*5
#define HROW 960      // hidden width:      256 + 128*3 + 64*5
#define TN   16       // nodes per workgroup tile
#define EPSN 1e-5f

// bf16 transposed+scaled weight layout inside d_ws (element offsets):
//  preT0 [256][128] @0        preT1 [128][64] @32768   preT2 [64][32] @40960
//  postT0[128][256] @43008    postT1[64][128] @75776   postT2[32][64] @83968
#define W_TOTAL 86016

__global__ __launch_bounds__(256)
void prep_weights(const float* pw0, const float* pw1, const float* pw2,
                  const float* qw0, const float* qw1, const float* qw2,
                  __bf16* ws)
{
    int idx = blockIdx.x * blockDim.x + threadIdx.x;
    if (idx >= W_TOTAL) return;
    const float* w; int mi, mo, off;
    if      (idx < 32768) { w = pw0; mi = 128; mo = 256; off = 0;     }
    else if (idx < 40960) { w = pw1; mi =  64; mo = 128; off = 32768; }
    else if (idx < 43008) { w = pw2; mi =  32; mo =  64; off = 40960; }
    else if (idx < 75776) { w = qw0; mi = 256; mo = 128; off = 43008; }
    else if (idx < 83968) { w = qw1; mi = 128; mo =  64; off = 75776; }
    else                  { w = qw2; mi =  64; mo =  32; off = 83968; }
    int rel = idx - off;
    int o = rel / mi;          // output channel (row of transposed matrix)
    int i = rel - o * mi;      // input channel  (contiguous -> B-frag K runs)
    float v = w[i * mo + o] * rsqrtf((float)mi);   // fold e3nn 1/sqrt(fan_in)
    ws[idx] = (__bf16)v;
}

// ---------------------------------------------------------------------------
// One 16x16 C-tile. A row / B col for this lane are fully resolved by the
// caller; element k of the row lives at arow[k] / brow[k] (contiguous).
// 16-bit fragment layout (ISA 7.12.2): lane half h holds K = ks*32 + h*8 + 0..7
// in v[0..3] and +16 in v[4..7] -> two 16-byte loads per fragment.
// C/D: col = lane&15, row = j + 8*(lane>=16); caller pre-resolves the column.
// ---------------------------------------------------------------------------
template<int KSTEPS>
__device__ __forceinline__ void gemm_tile(
    const __bf16* __restrict__ arow,   // LDS
    const __bf16* __restrict__ brow,   // global (L2-resident weights)
    float* __restrict__ ccol, int CROW, int half)
{
    v8f acc = {};
#pragma unroll
    for (int ks = 0; ks < KSTEPS; ++ks) {
        const __bf16* ap = arow + ks * 32 + half * 8;
        const __bf16* bp = brow + ks * 32 + half * 8;
        v8bf alo = *reinterpret_cast<const v8bf*>(ap);
        v8bf ahi = *reinterpret_cast<const v8bf*>(ap + 16);
        v8bf blo = *reinterpret_cast<const v8bf*>(bp);
        v8bf bhi = *reinterpret_cast<const v8bf*>(bp + 16);
        v16bf a = __builtin_shufflevector(alo, ahi,
                    0, 1, 2, 3, 4, 5, 6, 7, 8, 9, 10, 11, 12, 13, 14, 15);
        v16bf b = __builtin_shufflevector(blo, bhi,
                    0, 1, 2, 3, 4, 5, 6, 7, 8, 9, 10, 11, 12, 13, 14, 15);
        acc = __builtin_amdgcn_wmma_f32_16x16x32_bf16(
                  false, a, false, b, (short)0, acc, false, false);
    }
#pragma unroll
    for (int j = 0; j < 8; ++j)
        ccol[(j + half * 8) * CROW] = acc[j];
}

__global__ __launch_bounds__(128)
void ffn_kernel(const float* __restrict__ x,
                const float* __restrict__ nw0, const float* __restrict__ nb0,
                const float* __restrict__ nw1, const float* __restrict__ nw2,
                const __bf16* __restrict__ wbf,
                float* __restrict__ out, int N)
{
    // LDS layouts (per node row):
    //  ns : raw x, reference-interleaved; later reused as post-GEMM f32 output
    //  nsb: normalized bf16, m-major:  [0,128)=l0 | 128+m*64+c | 320+m*32+c
    //  hs : hidden f32,     m-major:   [0,256)=l0 | 256+m*128+o | 640+m*64+o
    //  hsb: gated hidden bf16, same m-major layout as hs
    __shared__ float  ns [TN * XROW];
    __shared__ __bf16 nsb[TN * XROW];
    __shared__ float  hs [TN * HROW];
    __shared__ __bf16 hsb[TN * HROW];
    __shared__ float  red[TN * 8 * 2];
    __shared__ float  stat[TN * 2];

    const int tid  = threadIdx.x;
    const int lane = tid & 31;
    // Wave id as an SGPR: makes the tile loops & dispatch scalar control flow.
    const int wave = __builtin_amdgcn_readfirstlane(tid >> 5);
    const int half = lane >> 4;          // K half selector for 16-bit fragments
    const int rn   = lane & 15;          // A row (node) / B col / C col index
    const long base_node = (long)blockIdx.x * TN;

    const __bf16* preT0  = wbf;
    const __bf16* preT1  = wbf + 32768;
    const __bf16* preT2  = wbf + 40960;
    const __bf16* postT0 = wbf + 43008;
    const __bf16* postT1 = wbf + 75776;
    const __bf16* postT2 = wbf + 83968;

    // ---- Phase 1: stage x tile into LDS -----------------------------------
    for (int i = tid; i < TN * XROW; i += blockDim.x) {
        long n = base_node + i / XROW;
        ns[i] = (n < N) ? x[(size_t)n * XROW + (i % XROW)] : 0.f;
    }
    __syncthreads();

    // ---- Phase 2: equivariant layernorm -> bf16 (m-major) into nsb --------
    {
        const int n = tid & 15, g = tid >> 4;          // 8 groups per node
        const float* xr = ns + n * XROW;
        __bf16* yr = nsb + n * XROW;

        // l=0: standard layernorm over 128 scalars
        float s = 0.f, s2 = 0.f;
        for (int c = g * 16; c < g * 16 + 16; ++c) { float v = xr[c]; s += v; s2 += v * v; }
        red[(n * 8 + g) * 2 + 0] = s;
        red[(n * 8 + g) * 2 + 1] = s2;
        __syncthreads();
        if (g == 0) {
            float ts = 0.f, ts2 = 0.f;
            for (int k = 0; k < 8; ++k) { ts += red[(n * 8 + k) * 2]; ts2 += red[(n * 8 + k) * 2 + 1]; }
            float mu = ts * (1.f / 128.f);
            float var = ts2 * (1.f / 128.f) - mu * mu;
            stat[n * 2 + 0] = mu;
            stat[n * 2 + 1] = rsqrtf(var + EPSN);
        }
        __syncthreads();
        {
            float mu = stat[n * 2], rs = stat[n * 2 + 1];
            for (int c = g * 16; c < g * 16 + 16; ++c)
                yr[c] = (__bf16)((xr[c] - mu) * rs * nw0[c] + nb0[c]);
        }
        __syncthreads();

        // l=1: RMS over 64 channel norms (3 components each)
        float sq = 0.f;
        for (int c = g * 8; c < g * 8 + 8; ++c) {
            const float* p = xr + 128 + c * 3;
            sq += p[0] * p[0] + p[1] * p[1] + p[2] * p[2];
        }
        red[n * 8 + g] = sq;
        __syncthreads();
        if (g == 0) {
            float ts = 0.f;
            for (int k = 0; k < 8; ++k) ts += red[n * 8 + k];
            stat[n * 2] = rsqrtf(ts * (1.f / 64.f) + EPSN);
        }
        __syncthreads();
        {
            float r1 = stat[n * 2];
            for (int c = g * 8; c < g * 8 + 8; ++c) {
                float w = nw1[c] * r1;
                const float* p = xr + 128 + c * 3;           // interleaved in
                yr[128 +   0 + c] = (__bf16)(p[0] * w);      // m-major out
                yr[128 +  64 + c] = (__bf16)(p[1] * w);
                yr[128 + 128 + c] = (__bf16)(p[2] * w);
            }
        }
        __syncthreads();

        // l=2: RMS over 32 channel norms (5 components each)
        float sq2 = 0.f;
        for (int c = g * 4; c < g * 4 + 4; ++c) {
            const float* p = xr + 320 + c * 5;
            for (int m = 0; m < 5; ++m) sq2 += p[m] * p[m];
        }
        red[n * 8 + g] = sq2;
        __syncthreads();
        if (g == 0) {
            float ts = 0.f;
            for (int k = 0; k < 8; ++k) ts += red[n * 8 + k];
            stat[n * 2] = rsqrtf(ts * (1.f / 32.f) + EPSN);
        }
        __syncthreads();
        {
            float r2 = stat[n * 2];
            for (int c = g * 4; c < g * 4 + 4; ++c) {
                float w = nw2[c] * r2;
                const float* p = xr + 320 + c * 5;
#pragma unroll
                for (int m = 0; m < 5; ++m)
                    yr[320 + m * 32 + c] = (__bf16)(p[m] * w);
            }
        }
    }
    __syncthreads();

    // ---- Phase 3: pre-linear 480->960 via bf16 WMMA -----------------------
    // 60 C-tiles: l0: 16 (K=128) | l1: 3(m) x 8 (K=64) | l2: 5(m) x 4 (K=32)
    for (int t = wave; t < 60; t += 4) {
        if (t < 16) {
            int oc = t * 16 + rn;
            gemm_tile<4>(nsb + rn * XROW,
                         preT0 + oc * 128,
                         hs + oc, HROW, half);
        } else if (t < 40) {
            int u = t - 16, m = u >> 3, oc = (u & 7) * 16 + rn;
            gemm_tile<2>(nsb + rn * XROW + 128 + m * 64,
                         preT1 + oc * 64,
                         hs + 256 + m * 128 + oc, HROW, half);
        } else {
            int u = t - 40, m = u >> 2, oc = (u & 3) * 16 + rn;
            gemm_tile<1>(nsb + rn * XROW + 320 + m * 32,
                         preT2 + oc * 32,
                         hs + 640 + m * 64 + oc, HROW, half);
        }
    }
    __syncthreads();

    // ---- Phase 4: gate -> bf16 into hsb (m-major) -------------------------
    for (int i = tid; i < TN * 256; i += blockDim.x) {          // l=0 sigmoid
        int n = i >> 8, c = i & 255;
        float v = hs[n * HROW + c];
        hsb[n * HROW + c] = (__bf16)(v / (1.f + __expf(-v)));
    }
    for (int i = tid; i < TN * 128; i += blockDim.x) {          // l=1 norm gate
        int n = i >> 7, c = i & 127;
        const float* p = hs + n * HROW + 256 + c;               // stride 128
        __bf16* q = hsb + n * HROW + 256 + c;
        float v0 = p[0], v1 = p[128], v2 = p[256];
        float nr = sqrtf(v0 * v0 + v1 * v1 + v2 * v2 + 1e-12f);
        float g = 1.f / (1.f + __expf(-nr));
        q[0] = (__bf16)(v0 * g); q[128] = (__bf16)(v1 * g); q[256] = (__bf16)(v2 * g);
    }
    for (int i = tid; i < TN * 64; i += blockDim.x) {           // l=2 norm gate
        int n = i >> 6, c = i & 63;
        const float* p = hs + n * HROW + 640 + c;               // stride 64
        __bf16* q = hsb + n * HROW + 640 + c;
        float v[5], s = 0.f;
#pragma unroll
        for (int m = 0; m < 5; ++m) { v[m] = p[m * 64]; s += v[m] * v[m]; }
        float g = 1.f / (1.f + __expf(-sqrtf(s + 1e-12f)));
#pragma unroll
        for (int m = 0; m < 5; ++m) q[m * 64] = (__bf16)(v[m] * g);
    }
    __syncthreads();

    // ---- Phase 5: post-linear 960->480 (f32 out into ns, interleaved) -----
    // 30 C-tiles: l0: 8 (K=256) | l1: 3(m) x 4 (K=128) | l2: 5(m) x 2 (K=64)
    for (int t = wave; t < 30; t += 4) {
        if (t < 8) {
            int oc = t * 16 + rn;
            gemm_tile<8>(hsb + rn * HROW,
                         postT0 + oc * 256,
                         ns + oc, XROW, half);
        } else if (t < 20) {
            int u = t - 8, m = u >> 2, oc = (u & 3) * 16 + rn;
            gemm_tile<4>(hsb + rn * HROW + 256 + m * 128,
                         postT1 + oc * 128,
                         ns + 128 + oc * 3 + m, XROW, half);    // o*3+m out
        } else {
            int u = t - 20, m = u >> 1, oc = (u & 1) * 16 + rn;
            gemm_tile<2>(hsb + rn * HROW + 640 + m * 64,
                         postT2 + oc * 64,
                         ns + 320 + oc * 5 + m, XROW, half);    // o*5+m out
        }
    }
    __syncthreads();

    // ---- Phase 6: residual add, store -------------------------------------
    for (int i = tid; i < TN * XROW; i += blockDim.x) {
        long n = base_node + i / XROW;
        if (n < N) {
            size_t gi = (size_t)n * XROW + (i % XROW);
            out[gi] = x[gi] + ns[i];
        }
    }
}

extern "C" void kernel_launch(void* const* d_in, const int* in_sizes, int n_in,
                              void* d_out, int out_size, void* d_ws, size_t ws_size,
                              hipStream_t stream)
{
    const float* x   = (const float*)d_in[0];
    const float* nw0 = (const float*)d_in[1];
    const float* nb0 = (const float*)d_in[2];
    const float* nw1 = (const float*)d_in[3];
    const float* nw2 = (const float*)d_in[4];
    const float* pw0 = (const float*)d_in[5];
    const float* pw1 = (const float*)d_in[6];
    const float* pw2 = (const float*)d_in[7];
    const float* qw0 = (const float*)d_in[8];
    const float* qw1 = (const float*)d_in[9];
    const float* qw2 = (const float*)d_in[10];

    __bf16* wbf = (__bf16*)d_ws;
    int N = in_sizes[0] / XROW;

    prep_weights<<<(W_TOTAL + 255) / 256, 256, 0, stream>>>(pw0, pw1, pw2, qw0, qw1, qw2, wbf);
    ffn_kernel<<<(N + TN - 1) / TN, 128, 0, stream>>>(x, nw0, nb0, nw1, nw2, wbf,
                                                      (float*)d_out, N);
}
